// MPPI_34849364639947
// MI455X (gfx1250) — matrix operand order repaired and checked by reading, not compiled, not run on hardware
//
#include <hip/hip_runtime.h>
#include <math.h>

#define DT       0.05f
#define LAMBDA   1.0f
#define UMINV   -2.0f
#define UMAXV    2.0f
#define Q_W      1.0f
#define R_W      0.1f

#define S_DIM   128
#define C_DIM   32
#define N_SAMP  16384
#define H_DIM   64
#define THR     13107   // int(16384 * 0.8)

typedef __attribute__((ext_vector_type(16))) __bf16        v16bf;
typedef __attribute__((ext_vector_type(8)))  float         v8f;
typedef __attribute__((ext_vector_type(8)))  unsigned int  v8u;

union FragU { uint4 u4[2]; v8u u8; v16bf bf; };

// Cheap f32->bf16 (round-half-up on raw bits; differs from RNE only at ties)
static __device__ inline unsigned short f2bf1(float x) {
  unsigned a = __builtin_bit_cast(unsigned, x);
  return (unsigned short)((a + 0x8000u) >> 16);
}
// pack two f32 -> {bf16(y):bf16(x)} in 3 VALU ops via v_perm_b32
static __device__ inline unsigned pk2bf(float x, float y) {
  unsigned a = __builtin_bit_cast(unsigned, x) + 0x8000u;
  unsigned b = __builtin_bit_cast(unsigned, y) + 0x8000u;
  return __builtin_amdgcn_perm(b, a, 0x07060302u);  // bytes {b3,b2,a3,a2}
}
static __device__ inline float clipf(float x) {
  return fminf(fmaxf(x, UMINV), UMAXV);   // v_med3_num_f32
}
static __device__ inline float fast_tanh(float x) {
#if __has_builtin(__builtin_amdgcn_tanhf)
  return __builtin_amdgcn_tanhf(x);       // v_tanh_f32 (CDNA5 trans op)
#else
  return tanhf(x);
#endif
}

// ---------------------------------------------------------------------------
// Kernel 0: convert A (S*S) and B (S*C) to bf16 in workspace
// ---------------------------------------------------------------------------
__global__ void mppi_convert_kernel(const float* __restrict__ A,
                                    const float* __restrict__ B,
                                    unsigned short* __restrict__ Abf,
                                    unsigned short* __restrict__ Bbf) {
  int i = blockIdx.x * blockDim.x + threadIdx.x;
  if (i < S_DIM * S_DIM) Abf[i] = f2bf1(A[i]);
  if (i < S_DIM * C_DIM) Bbf[i] = f2bf1(B[i]);
}

// ---------------------------------------------------------------------------
// Kernel 1: N rollouts + cost accumulation. 256 thr = 8 waves; wave = 16
// samples. 40 v_wmma_f32_16x16x32_bf16 per wave per timestep; f32 state in
// registers, bf16 mirror in LDS for A-fragment ds_load_b128s. No per-step
// barrier (per-wave LDS slice; DS ops are wave-ordered). Tile post-processing
// is software-pipelined one tile behind the WMMA issue, and next step's
// noise rows are prefetched before the tile loop.
// ---------------------------------------------------------------------------
__global__ __launch_bounds__(256)
void mppi_rollout_kernel(const float* __restrict__ state,
                         const unsigned short* __restrict__ Abf,
                         const unsigned short* __restrict__ Bbf,
                         const float* __restrict__ noises,
                         const float* __restrict__ prev,
                         float* __restrict__ costs) {
  __shared__ unsigned short Xs[8][16][S_DIM];   // bf16 state per wave: 32 KB
  __shared__ float costLds[128];

  const int tid    = threadIdx.x;
  const int w      = tid >> 5;
  const int L      = tid & 31;
  const int half   = L >> 4;        // 0 | 1
  const int lane16 = L & 15;
  const int rbase  = 8 * half;      // 2*rbase = K-base for A-frag; row-base for C/D frag
  const int n0     = blockIdx.x * 128 + w * 16;

  if (tid < 128) costLds[tid] = 0.0f;

  // lane-owned f32 state: xreg[j][r] = x[row r+rbase][col 16j+lane16]
  float xreg[8][8];
  float stAcc[8];
  float uAcc = 0.0f;
#pragma unroll
  for (int r = 0; r < 8; ++r) stAcc[r] = 0.0f;
#pragma unroll
  for (int j = 0; j < 8; ++j) {
    float x0 = state[16 * j + lane16];
#pragma unroll
    for (int r = 0; r < 8; ++r) xreg[j][r] = x0;
  }
  // stage cost includes ||x0||^2 once
#pragma unroll
  for (int j = 0; j < 8; ++j) {
    float x0 = xreg[j][0];
#pragma unroll
    for (int r = 0; r < 8; ++r) stAcc[r] += x0 * x0;
  }
  // seed LDS bf16 state (each lane writes its owned (row,col) elements)
#pragma unroll
  for (int j = 0; j < 8; ++j) {
    unsigned short b = f2bf1(xreg[j][0]);
#pragma unroll
    for (int r = 0; r < 8; ++r) Xs[w][r + rbase][16 * j + lane16] = b;
  }

  const int   myN  = n0 + lane16;
  const float mInh = (myN < THR) ? 1.0f : 0.0f;
  const float* nbase = noises + (size_t)myN * H_DIM * C_DIM;

  // prefetch t=0 noise/prev rows
  float4 na[4], pa[4];
  {
    const float* nrow = nbase;
    const float* prow = prev;
    na[0] = *(const float4*)(nrow + 2 * rbase);
    na[1] = *(const float4*)(nrow + 2 * rbase + 4);
    na[2] = *(const float4*)(nrow + 2 * rbase + 8);
    na[3] = *(const float4*)(nrow + 2 * rbase + 12);
    pa[0] = *(const float4*)(prow + 2 * rbase);
    pa[1] = *(const float4*)(prow + 2 * rbase + 4);
    pa[2] = *(const float4*)(prow + 2 * rbase + 8);
    pa[3] = *(const float4*)(prow + 2 * rbase + 12);
  }

  for (int t = 0; t < H_DIM; ++t) {
    // A-fragments of X (16x32 bf16 per K-tile): two ds_load_b128 each
    FragU af[4];
#pragma unroll
    for (int kk = 0; kk < 4; ++kk) {
      const unsigned short* p = &Xs[w][lane16][32 * kk + 2 * rbase];
      af[kk].u4[0] = *(const uint4*)p;          // K = 16h .. 16h+7
      af[kk].u4[1] = *(const uint4*)(p + 8);    // K = 16h+8 .. 16h+15
    }

    // U fragment from prefetched rows: clip(noise + m*prev); R*u^2 accum
    FragU uf;
    {
      const float* nv = (const float*)na;
      const float* pv = (const float*)pa;
#pragma unroll
      for (int q = 0; q < 8; ++q) {
        float u0 = clipf(fmaf(mInh, pv[2 * q],     nv[2 * q]));
        float u1 = clipf(fmaf(mInh, pv[2 * q + 1], nv[2 * q + 1]));
        uAcc = fmaf(u0, u0, uAcc);
        uAcc = fmaf(u1, u1, uAcc);
        uf.u8[q] = pk2bf(u0, u1);
      }
    }

    // prefetch next step's noise/prev rows (hidden behind the tile loop)
    if (t + 1 < H_DIM) {
      const float* nrow = nbase + (t + 1) * C_DIM;
      const float* prow = prev + (t + 1) * C_DIM;
      na[0] = *(const float4*)(nrow + 2 * rbase);
      na[1] = *(const float4*)(nrow + 2 * rbase + 4);
      na[2] = *(const float4*)(nrow + 2 * rbase + 8);
      na[3] = *(const float4*)(nrow + 2 * rbase + 12);
      pa[0] = *(const float4*)(prow + 2 * rbase);
      pa[1] = *(const float4*)(prow + 2 * rbase + 4);
      pa[2] = *(const float4*)(prow + 2 * rbase + 8);
      pa[3] = *(const float4*)(prow + 2 * rbase + 12);
    }

    // 5 WMMAs for one 16x16 output tile: Y = X@A^T + U@B^T
    auto tileWmma = [&](int j) -> v8f {
      v8f acc = {0.f, 0.f, 0.f, 0.f, 0.f, 0.f, 0.f, 0.f};
      const int orow = 16 * j + lane16;
      const unsigned short* arow = Abf + (size_t)orow * S_DIM + 16 * half;
#pragma unroll
      for (int kk = 0; kk < 4; ++kk) {
        FragU bfr;
        bfr.u8 = *(const v8u*)(arow + 32 * kk);  // A[orow][32kk+16h .. +15]
        acc = __builtin_amdgcn_wmma_f32_16x16x32_bf16(
            false, af[kk].bf, false, bfr.bf, (short)0, acc, false, false);
      }
      FragU bbr;
      bbr.u8 = *(const v8u*)(Bbf + (size_t)orow * C_DIM + 16 * half);
      return __builtin_amdgcn_wmma_f32_16x16x32_bf16(
          false, uf.bf, false, bbr.bf, (short)0, acc, false, false);
    };
    // x' = x + dt*tanh(Y); cost accum; bf16 write-back to LDS
    auto postProc = [&](int j, v8f acc) {
      const int orow = 16 * j + lane16;
#pragma unroll
      for (int r = 0; r < 8; ++r) {
        float xn = fmaf(DT, fast_tanh(acc[r]), xreg[j][r]);
        stAcc[r] = fmaf(xn, xn, stAcc[r]);
        xreg[j][r] = xn;
        Xs[w][r + rbase][orow] = f2bf1(xn);
      }
    };

    // software pipeline: WMMA tile j while post-processing tile j-1
    v8f accP = tileWmma(0);
#pragma unroll
    for (int j = 1; j < 8; ++j) {
      v8f accC = tileWmma(j);
      postProc(j - 1, accP);
      accP = accC;
    }
    postProc(7, accP);
  }

  __syncthreads();   // covers costLds init + all waves done
#pragma unroll
  for (int r = 0; r < 8; ++r)
    atomicAdd(&costLds[w * 16 + r + rbase], Q_W * stAcc[r]);
  atomicAdd(&costLds[w * 16 + lane16], R_W * uAcc);
  __syncthreads();
  if (tid < 128) costs[blockIdx.x * 128 + tid] = costLds[tid];
}

// ---------------------------------------------------------------------------
// Kernel 2: softmax over N costs -> weights (single workgroup, 64 KB of data)
// ---------------------------------------------------------------------------
__global__ __launch_bounds__(256)
void mppi_softmax_kernel(const float* __restrict__ costs,
                         float* __restrict__ weights) {
  __shared__ float red[256];
  const int tid = threadIdx.x;
  float mn = 3.4e38f;
  for (int n = tid; n < N_SAMP; n += 256) mn = fminf(mn, costs[n]);
  red[tid] = mn; __syncthreads();
  for (int s = 128; s > 0; s >>= 1) {
    if (tid < s) red[tid] = fminf(red[tid], red[tid + s]);
    __syncthreads();
  }
  const float minc = red[0];
  __syncthreads();
  float sum = 0.0f;
  for (int n = tid; n < N_SAMP; n += 256)
    sum += expf(-(costs[n] - minc) * (1.0f / LAMBDA));
  red[tid] = sum; __syncthreads();
  for (int s = 128; s > 0; s >>= 1) {
    if (tid < s) red[tid] += red[tid + s];
    __syncthreads();
  }
  const float inv = 1.0f / red[0];
  for (int n = tid; n < N_SAMP; n += 256)
    weights[n] = expf(-(costs[n] - minc) * (1.0f / LAMBDA)) * inv;
}

// ---------------------------------------------------------------------------
// Kernel 3: optimal_action[h][c] = sum_n w_n * clip(noise + prev*(n<thr))
// One WG per h; 32-lane groups read contiguous 128 B rows (coalesced).
// ---------------------------------------------------------------------------
__global__ __launch_bounds__(256)
void mppi_wsum_kernel(const float* __restrict__ noises,
                      const float* __restrict__ prev,
                      const float* __restrict__ weights,
                      float* __restrict__ outAct) {
  __shared__ float red[256];
  const int h   = blockIdx.x;
  const int tid = threadIdx.x;
  const int c   = tid & 31;
  const int g   = tid >> 5;
  const float pv = prev[h * C_DIM + c];
  float acc = 0.0f;
  for (int n = g; n < N_SAMP; n += 8) {
    float u = noises[((size_t)n * H_DIM + h) * C_DIM + c];
    u += (n < THR) ? pv : 0.0f;
    u = clipf(u);
    acc = fmaf(weights[n], u, acc);
  }
  red[tid] = acc; __syncthreads();
  if (g == 0) {
#pragma unroll
    for (int gg = 1; gg < 8; ++gg) acc += red[gg * 32 + c];
    outAct[h * C_DIM + c] = acc;
  }
}

// ---------------------------------------------------------------------------
// Kernel 4: final rollout of optimal action sequence (tiny; f32 exact)
// ---------------------------------------------------------------------------
__global__ __launch_bounds__(128)
void mppi_final_rollout_kernel(const float* __restrict__ state,
                               const float* __restrict__ A,
                               const float* __restrict__ B,
                               const float* __restrict__ uopt,
                               float* __restrict__ outState) {
  __shared__ float xc[S_DIM];
  const int i = threadIdx.x;
  float xi = state[i];
  xc[i] = xi;
  outState[i] = xi;
  const float* Arow = A + (size_t)i * S_DIM;
  const float* Brow = B + (size_t)i * C_DIM;
  for (int t = 0; t < H_DIM; ++t) {
    __syncthreads();
    float acc = 0.0f;
    for (int k = 0; k < S_DIM; ++k) acc = fmaf(Arow[k], xc[k], acc);
    const float* u = uopt + t * C_DIM;
    for (int cc = 0; cc < C_DIM; ++cc) acc = fmaf(Brow[cc], u[cc], acc);
    float xn = xi + DT * tanhf(acc);
    __syncthreads();
    xc[i] = xn;
    xi = xn;
    outState[(t + 1) * S_DIM + i] = xn;
  }
}

// ---------------------------------------------------------------------------
extern "C" void kernel_launch(void* const* d_in, const int* in_sizes, int n_in,
                              void* d_out, int out_size, void* d_ws, size_t ws_size,
                              hipStream_t stream) {
  const float* state  = (const float*)d_in[0];   // [128]
  const float* A      = (const float*)d_in[1];   // [128,128]
  const float* B      = (const float*)d_in[2];   // [128,32]
  const float* noises = (const float*)d_in[3];   // [16384,64,32]
  const float* prev   = (const float*)d_in[4];   // [64,32]

  float* outAct   = (float*)d_out;                      // [H,C] = 2048
  float* outState = (float*)d_out + H_DIM * C_DIM;      // [1,H+1,S] = 8320

  char* ws = (char*)d_ws;
  unsigned short* Abf   = (unsigned short*)ws;                      // 32768 B
  unsigned short* Bbf   = (unsigned short*)(ws + 32768);            //  8192 B
  float*          costs = (float*)(ws + 40960);                     // 65536 B
  float*          wts   = (float*)(ws + 40960 + 65536);             // 65536 B

  mppi_convert_kernel<<<(S_DIM * S_DIM + 255) / 256, 256, 0, stream>>>(A, B, Abf, Bbf);
  mppi_rollout_kernel<<<N_SAMP / 128, 256, 0, stream>>>(state, Abf, Bbf, noises, prev, costs);
  mppi_softmax_kernel<<<1, 256, 0, stream>>>(costs, wts);
  mppi_wsum_kernel<<<H_DIM, 256, 0, stream>>>(noises, prev, wts, outAct);
  mppi_final_rollout_kernel<<<1, S_DIM, 0, stream>>>(state, A, B, outAct, outState);
}
